// EncoderDecoder_88828513616260
// MI455X (gfx1250) — compile-verified, matrix operand order checked by reading, not run on hardware
//
#include <hip/hip_runtime.h>
#include <hip/hip_bf16.h>
#include <stdint.h>

typedef __attribute__((ext_vector_type(16))) __bf16 v16bf;
typedef __attribute__((ext_vector_type(8)))  float  v8f;

#define HID    512
#define NBATCH 2048
#define INSZ   128
#define OUTSZ  128

union FragU { v16bf v; uint4 u[2]; };

// Load a 16x32 bf16 fragment from LDS (row-major, ld = 32 elements).
// ISA layout (16-bit A/B): lanes 0-15 hold K={0..7,16..23}, lanes 16-31 hold
// K={8..15,24..31} of row (rowBase+half) -> two 16-byte contiguous loads.
__device__ __forceinline__ v16bf ldsFrag(const __bf16* lds, int rowBase, int half, int hi) {
    const uint4* p = reinterpret_cast<const uint4*>(lds) + (size_t)(rowBase + half) * 4 + hi;
    FragU f;
    f.u[0] = p[0];   // K = hi*8 .. hi*8+7
    f.u[1] = p[2];   // K = 16+hi*8 .. 16+hi*8+7
    return f.v;
}

__device__ __forceinline__ float fast_sigmoid(float x) { return 1.0f / (1.0f + __expf(-x)); }
__device__ __forceinline__ float fast_tanh(float x)    { return 2.0f / (1.0f + __expf(-2.0f * x)) - 1.0f; }

// gfx1250 async global->LDS copy, 16B per lane, tracked by ASYNCcnt.
__device__ __forceinline__ void async_ld_b128(unsigned ldsOff, const void* g) {
    asm volatile("global_load_async_to_lds_b128 %0, %1, off"
                 :: "v"(ldsOff), "v"(g) : "memory");
}
// Wait until at most n async loads remain outstanding (exact immediates).
__device__ __forceinline__ void wait_async_keep(int n) {
    switch (n) {
    case 0:  asm volatile("s_wait_asynccnt 0" ::: "memory"); break;
    case 2:  asm volatile("s_wait_asynccnt 2" ::: "memory"); break;
    case 4:  asm volatile("s_wait_asynccnt 4" ::: "memory"); break;
    default: asm volatile("s_wait_asynccnt 6" ::: "memory"); break;
    }
}

__device__ __forceinline__ unsigned ldsAddr(const void* p) {
    return (unsigned)(uintptr_t)p;   // low 32 bits of flat shared ptr = LDS offset
}

// One LSTM cell step: gates = A1 @ W1^T + A2 @ W2^T + bias; update c, h.
// Tile: 128 batch rows x 64 hidden cols x 4 gates. Grid (16, 8), block 256.
// Double-buffered LDS, async global->LDS staging.
template<bool A1F32, int K1>
__global__ __launch_bounds__(256) void lstm_step_kernel(
    const float*  __restrict__ A1f,   // f32 input (encoder x_t) or unused
    const __bf16* __restrict__ A1h,   // bf16 input (decoder h) or unused
    const __bf16* __restrict__ W1,    // [4*HID, K1] bf16
    const __bf16* __restrict__ A2h,   // h_in bf16 [NBATCH, HID]
    const __bf16* __restrict__ W2,    // [4*HID, HID] bf16
    const float*  __restrict__ bias,  // [4*HID] fused b_ih + b_hh
    float* __restrict__ cbuf,         // [NBATCH, HID] fp32 cell state (in/out)
    float* __restrict__ hf,           // [NBATCH, HID] fp32 h out
    __bf16* __restrict__ hb)          // [NBATCH, HID] bf16 h out (ping-pong)
{
    constexpr int NC1 = K1 / 32;
    constexpr int NC2 = HID / 32;
    constexpr int NC  = NC1 + NC2;

    __shared__ __align__(16) __bf16 aLds[2][128 * 32];
    __shared__ __align__(16) __bf16 bLds[2][256 * 32];

    const int tid  = threadIdx.x;
    const int lane = tid & 31;
    const int w    = tid >> 5;
    const int half = lane & 15;
    const int hi   = lane >> 4;
    const int mr   = w & 3;     // row block (32 rows each)
    const int wc   = w >> 2;    // column half (32 cols each)
    const int row0 = blockIdx.x * 128;
    const int n0   = blockIdx.y * 64;

    // Number of async loads issued by stage(c) (per thread / per wave).
    auto issued = [&](int c) -> int { return (A1F32 && c < NC1) ? 4 : 6; };

    // Stage chunk c into LDS buffer `buf`.
    auto stage = [&](int c, int buf) {
        const __bf16* W;
        int ldW, kb;
        bool seg1 = (c < NC1);
        if (seg1) { W = W1; ldW = K1;  kb = c * 32; }
        else      { W = W2; ldW = HID; kb = (c - NC1) * 32; }

        // B tile: 256 rows (4 gates x 64 cols), one W row per thread, 4x b128.
        {
            const __bf16* src = W + (size_t)((tid >> 6) * HID + n0 + (tid & 63)) * ldW + kb;
            unsigned dst = ldsAddr(bLds[buf] + (size_t)tid * 32);
            async_ld_b128(dst,      src);
            async_ld_b128(dst + 16, src + 8);
            async_ld_b128(dst + 32, src + 16);
            async_ld_b128(dst + 48, src + 24);
        }
        // A tile: 128 rows x 32. thread -> row tid>>1, k-half (tid&1)*16.
        const int r  = tid >> 1;
        const int kq = (tid & 1) << 4;
        if (A1F32 && seg1) {
            // fp32 source: sync load + convert + ds_store (encoder x_t only).
            const float4* s = reinterpret_cast<const float4*>(
                A1f + (size_t)(row0 + r) * K1 + kb + kq);
            float4 f0 = s[0], f1 = s[1], f2 = s[2], f3 = s[3];
            union { __bf16 t[16]; uint4 u[2]; } tmp;
            tmp.t[0]  = (__bf16)f0.x; tmp.t[1]  = (__bf16)f0.y;
            tmp.t[2]  = (__bf16)f0.z; tmp.t[3]  = (__bf16)f0.w;
            tmp.t[4]  = (__bf16)f1.x; tmp.t[5]  = (__bf16)f1.y;
            tmp.t[6]  = (__bf16)f1.z; tmp.t[7]  = (__bf16)f1.w;
            tmp.t[8]  = (__bf16)f2.x; tmp.t[9]  = (__bf16)f2.y;
            tmp.t[10] = (__bf16)f2.z; tmp.t[11] = (__bf16)f2.w;
            tmp.t[12] = (__bf16)f3.x; tmp.t[13] = (__bf16)f3.y;
            tmp.t[14] = (__bf16)f3.z; tmp.t[15] = (__bf16)f3.w;
            uint4* d = reinterpret_cast<uint4*>(aLds[buf] + (size_t)r * 32 + kq);
            d[0] = tmp.u[0]; d[1] = tmp.u[1];
        } else {
            const __bf16* Ah = seg1 ? A1h : A2h;
            const int ldA    = seg1 ? K1 : HID;
            const __bf16* src = Ah + (size_t)(row0 + r) * ldA + kb + kq;
            unsigned dst = ldsAddr(aLds[buf] + (size_t)r * 32 + kq);
            async_ld_b128(dst,      src);
            async_ld_b128(dst + 16, src + 8);
        }
    };

    v8f acc[4][2][2];
#pragma unroll
    for (int g = 0; g < 4; ++g)
#pragma unroll
        for (int c = 0; c < 2; ++c)
#pragma unroll
            for (int a = 0; a < 2; ++a)
#pragma unroll
                for (int j = 0; j < 8; ++j) acc[g][c][a][j] = 0.0f;

    stage(0, 0);
    int cur = 0;
    for (int c = 0; c < NC; ++c) {
        if (c + 1 < NC) stage(c + 1, cur ^ 1);             // prefetch next chunk
        wait_async_keep((c + 1 < NC) ? issued(c + 1) : 0); // drain current chunk
        __syncthreads();                                   // publish LDS writes

        const __bf16* aB = aLds[cur];
        const __bf16* bB = bLds[cur];
        v16bf a0 = ldsFrag(aB, 32 * mr,      half, hi);
        v16bf a1 = ldsFrag(aB, 32 * mr + 16, half, hi);
#pragma unroll
        for (int g = 0; g < 4; ++g)
#pragma unroll
            for (int cf = 0; cf < 2; ++cf) {
                v16bf b = ldsFrag(bB, g * 64 + wc * 32 + cf * 16, half, hi);
                acc[g][cf][0] = __builtin_amdgcn_wmma_f32_16x16x32_bf16(
                    false, a0, false, b, (short)0, acc[g][cf][0], false, false);
                acc[g][cf][1] = __builtin_amdgcn_wmma_f32_16x16x32_bf16(
                    false, a1, false, b, (short)0, acc[g][cf][1], false, false);
            }
        __syncthreads();        // all reads of buf `cur` done before re-stage
        cur ^= 1;
    }

    // ---- Epilogue: LSTM activations + state update ----
#pragma unroll
    for (int cf = 0; cf < 2; ++cf) {
        const int col = n0 + wc * 32 + cf * 16 + half;
        const float bi  = bias[col];
        const float bff = bias[HID + col];
        const float bg  = bias[2 * HID + col];
        const float bo  = bias[3 * HID + col];
#pragma unroll
        for (int ar = 0; ar < 2; ++ar)
#pragma unroll
            for (int r = 0; r < 8; ++r) {
                const int row = row0 + 32 * mr + 16 * ar + r + 8 * hi;
                const size_t idx = (size_t)row * HID + col;
                float iv = fast_sigmoid(acc[0][cf][ar][r] + bi);
                float fv = fast_sigmoid(acc[1][cf][ar][r] + bff);
                float gv = fast_tanh(acc[2][cf][ar][r] + bg);
                float ov = fast_sigmoid(acc[3][cf][ar][r] + bo);
                float cn = fv * cbuf[idx] + iv * gv;
                cbuf[idx] = cn;
                float hv = ov * fast_tanh(cn);
                hf[idx] = hv;
                hb[idx] = (__bf16)hv;
            }
    }
}

// out[2048,128] = h_bf16 @ W_lin^T + b_lin.  Grid: (32, 2), block 256.
// Double-buffered LDS with async staging (2 async b128 per thread per chunk).
__global__ __launch_bounds__(256) void proj_kernel(
    const __bf16* __restrict__ A,     // [NBATCH, HID] bf16
    const __bf16* __restrict__ Wl,    // [OUTSZ, HID] bf16
    const float*  __restrict__ blin,  // [OUTSZ]
    float* __restrict__ outp)         // [NBATCH, OUTSZ]
{
    __shared__ __align__(16) __bf16 aLds[2][64 * 32];
    __shared__ __align__(16) __bf16 bLds[2][64 * 32];

    const int tid  = threadIdx.x;
    const int lane = tid & 31;
    const int w    = tid >> 5;
    const int half = lane & 15;
    const int hi   = lane >> 4;
    const int mr   = w & 3;
    const int wc   = w >> 2;
    const int row0 = blockIdx.x * 64;
    const int n0   = blockIdx.y * 64;

    constexpr int NC = HID / 32;   // 16 chunks

    auto stage = [&](int c, int buf) {
        const int r  = tid >> 2;           // 0..63
        const int kq = (tid & 3) << 3;     // 0,8,16,24
        const int kb = c * 32;
        async_ld_b128(ldsAddr(aLds[buf] + r * 32 + kq),
                      A  + (size_t)(row0 + r) * HID + kb + kq);
        async_ld_b128(ldsAddr(bLds[buf] + r * 32 + kq),
                      Wl + (size_t)(n0 + r) * HID + kb + kq);
    };

    v8f acc[2];
#pragma unroll
    for (int c = 0; c < 2; ++c)
#pragma unroll
        for (int j = 0; j < 8; ++j) acc[c][j] = 0.0f;

    stage(0, 0);
    int cur = 0;
    for (int c = 0; c < NC; ++c) {
        if (c + 1 < NC) stage(c + 1, cur ^ 1);
        wait_async_keep((c + 1 < NC) ? 2 : 0);
        __syncthreads();

        v16bf a = ldsFrag(aLds[cur], 16 * mr, half, hi);
#pragma unroll
        for (int cf = 0; cf < 2; ++cf) {
            v16bf b = ldsFrag(bLds[cur], wc * 32 + cf * 16, half, hi);
            acc[cf] = __builtin_amdgcn_wmma_f32_16x16x32_bf16(
                false, a, false, b, (short)0, acc[cf], false, false);
        }
        __syncthreads();
        cur ^= 1;
    }

#pragma unroll
    for (int cf = 0; cf < 2; ++cf) {
        const int col = n0 + wc * 32 + cf * 16 + half;
        const float bb = blin[col];
#pragma unroll
        for (int r = 0; r < 8; ++r) {
            const int row = row0 + 16 * mr + r + 8 * hi;
            outp[(size_t)row * OUTSZ + col] = acc[cf][r] + bb;
        }
    }
}

// One-time prep: weights -> bf16, fuse biases, init h/c state.
__global__ void prep_kernel(
    const float* __restrict__ Wih_e, const float* __restrict__ Whh_e,
    const float* __restrict__ Wih_d, const float* __restrict__ Whh_d,
    const float* __restrict__ Wlin,
    const float* __restrict__ bih_e, const float* __restrict__ bhh_e,
    const float* __restrict__ bih_d, const float* __restrict__ bhh_d,
    const float* __restrict__ h0,    const float* __restrict__ c0,
    __bf16* wih_e_h, __bf16* whh_e_h, __bf16* wih_d_h, __bf16* whh_d_h,
    __bf16* wlin_h, float* bias_e, float* bias_d,
    __bf16* hb, float* hf, float* cbuf)
{
    size_t i0 = (size_t)blockIdx.x * blockDim.x + threadIdx.x;
    size_t stride = (size_t)gridDim.x * blockDim.x;
    for (size_t i = i0; i < 1048576; i += stride) {
        whh_e_h[i] = (__bf16)Whh_e[i];
        wih_d_h[i] = (__bf16)Wih_d[i];
        whh_d_h[i] = (__bf16)Whh_d[i];
        if (i < 262144) wih_e_h[i] = (__bf16)Wih_e[i];
        if (i < 65536)  wlin_h[i]  = (__bf16)Wlin[i];
        if (i < 2048) {
            bias_e[i] = bih_e[i] + bhh_e[i];
            bias_d[i] = bih_d[i] + bhh_d[i];
        }
        float h = h0[i];
        hb[i]   = (__bf16)h;
        hf[i]   = h;
        cbuf[i] = c0[i];
    }
}

// After encoder: emit h_enc / c_enc, re-init c from cell_decoder.
__global__ void mid_kernel(const float* __restrict__ hf, const float* __restrict__ cbuf,
                           const float* __restrict__ c_dec0,
                           float* h_enc_out, float* c_enc_out, float* c_new)
{
    size_t i0 = (size_t)blockIdx.x * blockDim.x + threadIdx.x;
    size_t stride = (size_t)gridDim.x * blockDim.x;
    for (size_t i = i0; i < 1048576; i += stride) {
        h_enc_out[i] = hf[i];
        c_enc_out[i] = cbuf[i];
        c_new[i]     = c_dec0[i];
    }
}

__global__ void fin_kernel(const float* __restrict__ hf, const float* __restrict__ cbuf,
                           float* h_dec_out, float* c_dec_out)
{
    size_t i0 = (size_t)blockIdx.x * blockDim.x + threadIdx.x;
    size_t stride = (size_t)gridDim.x * blockDim.x;
    for (size_t i = i0; i < 1048576; i += stride) {
        h_dec_out[i] = hf[i];
        c_dec_out[i] = cbuf[i];
    }
}

extern "C" void kernel_launch(void* const* d_in, const int* in_sizes, int n_in,
                              void* d_out, int out_size, void* d_ws, size_t ws_size,
                              hipStream_t stream) {
    (void)in_sizes; (void)n_in; (void)out_size; (void)ws_size;

    const float* input    = (const float*)d_in[0];
    const float* hid_enc0 = (const float*)d_in[1];
    /* d_in[2] hidden_decoder: unused by the reference computation */
    const float* cell_enc0 = (const float*)d_in[3];
    const float* cell_dec0 = (const float*)d_in[4];
    const float* Wih_e = (const float*)d_in[5];
    const float* Whh_e = (const float*)d_in[6];
    const float* bih_e = (const float*)d_in[7];
    const float* bhh_e = (const float*)d_in[8];
    const float* Wih_d = (const float*)d_in[9];
    const float* Whh_d = (const float*)d_in[10];
    const float* bih_d = (const float*)d_in[11];
    const float* bhh_d = (const float*)d_in[12];
    const float* Wlin  = (const float*)d_in[13];
    const float* blin  = (const float*)d_in[14];
    float* out = (float*)d_out;

    // ---- workspace carve-up (~20 MB) ----
    char* ws = (char*)d_ws;
    size_t off = 0;
    auto alloc = [&](size_t bytes) -> void* {
        void* p = ws + off;
        off += (bytes + 255) & ~(size_t)255;
        return p;
    };
    __bf16* wih_e_h = (__bf16*)alloc((size_t)262144 * 2);
    __bf16* whh_e_h = (__bf16*)alloc((size_t)1048576 * 2);
    __bf16* wih_d_h = (__bf16*)alloc((size_t)1048576 * 2);
    __bf16* whh_d_h = (__bf16*)alloc((size_t)1048576 * 2);
    __bf16* wlin_h  = (__bf16*)alloc((size_t)65536 * 2);
    float*  bias_e  = (float*)alloc((size_t)2048 * 4);
    float*  bias_d  = (float*)alloc((size_t)2048 * 4);
    __bf16* hbuf[2];
    hbuf[0] = (__bf16*)alloc((size_t)1048576 * 2);
    hbuf[1] = (__bf16*)alloc((size_t)1048576 * 2);
    float* hf   = (float*)alloc((size_t)1048576 * 4);
    float* cbuf = (float*)alloc((size_t)1048576 * 4);

    prep_kernel<<<512, 256, 0, stream>>>(Wih_e, Whh_e, Wih_d, Whh_d, Wlin,
                                         bih_e, bhh_e, bih_d, bhh_d,
                                         hid_enc0, cell_enc0,
                                         wih_e_h, whh_e_h, wih_d_h, whh_d_h,
                                         wlin_h, bias_e, bias_d,
                                         hbuf[0], hf, cbuf);

    dim3 blk(256);
    dim3 gStep(NBATCH / 128, HID / 64);  // 16 x 8
    dim3 gProj(NBATCH / 64, OUTSZ / 64); // 32 x 2

    int p = 0;
    // Encoder: 100 sequential steps, x_t fp32 converted during staging.
    for (int t = 0; t < 100; ++t) {
        lstm_step_kernel<true, INSZ><<<gStep, blk, 0, stream>>>(
            input + (size_t)t * NBATCH * INSZ, nullptr, wih_e_h,
            hbuf[p], whh_e_h, bias_e, cbuf, hf, hbuf[1 - p]);
        p ^= 1;
    }

    float* h_enc_out = out + (size_t)30 * NBATCH * OUTSZ;
    float* h_dec_out = h_enc_out + (size_t)NBATCH * HID;
    float* c_enc_out = h_dec_out + (size_t)NBATCH * HID;
    float* c_dec_out = c_enc_out + (size_t)NBATCH * HID;

    mid_kernel<<<512, 256, 0, stream>>>(hf, cbuf, cell_dec0, h_enc_out, c_enc_out, cbuf);

    // output[0] = h_enc @ W_lin^T + b_lin
    proj_kernel<<<gProj, blk, 0, stream>>>(hbuf[p], wlin_h, blin, out);

    // Decoder: x = h (both GEMM inputs are the current hidden state).
    for (int s = 0; s < 29; ++s) {
        lstm_step_kernel<false, HID><<<gStep, blk, 0, stream>>>(
            nullptr, hbuf[p], wih_d_h,
            hbuf[p], whh_d_h, bias_d, cbuf, hf, hbuf[1 - p]);
        p ^= 1;
        proj_kernel<<<gProj, blk, 0, stream>>>(hbuf[p], wlin_h, blin,
                                               out + (size_t)(s + 1) * NBATCH * OUTSZ);
    }

    fin_kernel<<<512, 256, 0, stream>>>(hf, cbuf, h_dec_out, c_dec_out);
}
